// LTC_81166291959854
// MI455X (gfx1250) — matrix operand order repaired and checked by reading, not compiled, not run on hardware
//
#include <hip/hip_runtime.h>
#include <hip/hip_bf16.h>

typedef __attribute__((ext_vector_type(16))) __bf16 v16bf;
typedef __attribute__((ext_vector_type(8)))  __bf16 v8bf;
typedef __attribute__((ext_vector_type(8)))  float  v8f;

#define T_STEPS 256
#define BATCH   256
#define NN      512     // N (output features)
#define KK      1024    // 2N (concat input)
#define TB      16      // batch rows per workgroup (= WMMA M)
#define KT_TILES 32     // KK / 32
#define THREADS  512    // 16 waves (wave32)
#define TILES_PER_WAVE 2  // 32 N-tiles / 16 waves

// ---------------------------------------------------------------------------
// Weight conversion: fp32 W[N][2N] (GEMM uses B = W^T, B[k][n] = W[n][k])
// -> bf16, WMMA-B-fragment-major:  frag[nt][kt][lane][j]
//    lane l (h = l>>4, ln = l&15), element j holds  B[kt*32 + (j<8 ? 8*h + j : 16 + 8*h + j - 8)][nt*16 + ln]
// so the main kernel loads one contiguous 32-byte chunk per lane per fragment.
// ---------------------------------------------------------------------------
__global__ void wconvert_kernel(const float* __restrict__ W, __bf16* __restrict__ out) {
    int idx = blockIdx.x * blockDim.x + threadIdx.x;   // 0 .. NN*KK-1
    int j  = idx & 15;
    int l  = (idx >> 4) & 31;
    int kt = (idx >> 9) & 31;
    int nt = idx >> 14;
    int n = nt * 16 + (l & 15);
    int k = kt * 32 + (j < 8 ? 8 * (l >> 4) + j : 16 + 8 * (l >> 4) + (j - 8));
    out[idx] = (__bf16)W[(size_t)n * KK + k];
}

// Async DMA of 64 bytes/lane (whole slab = THREADS*64B = 32KB) global -> LDS.
// Generic LDS addresses carry the LDS byte offset in their low 32 bits.
__device__ __forceinline__ void async_slab_load(const float* gsrc, float* lds_dst, int tid) {
    unsigned long long ga = (unsigned long long)(const void*)(gsrc + tid * 16);
    unsigned la = (unsigned)(unsigned long long)(const void*)(lds_dst + tid * 16);
    asm volatile(
        "global_load_async_to_lds_b128 %0, %1, off\n\t"
        "global_load_async_to_lds_b128 %0, %1, off offset:16\n\t"
        "global_load_async_to_lds_b128 %0, %1, off offset:32\n\t"
        "global_load_async_to_lds_b128 %0, %1, off offset:48"
        :: "v"(la), "v"(ga) : "memory");
}

__device__ __forceinline__ void wait_async0() {
    asm volatile("s_wait_asynccnt 0x0" ::: "memory");
}

// Build a 16x32 bf16 A fragment per the ISA layout: lane half h holds
// K = 8h+0..7 (first v8bf) and K = 16+8h+0..7 (v8bf at +16 elements).
__device__ __forceinline__ v16bf a_frag(const __bf16* row) {
    const v8bf* p = (const v8bf*)row;
    v8bf lo = p[0], hi = p[2];
    return __builtin_shufflevector(lo, hi, 0, 1, 2, 3, 4, 5, 6, 7,
                                           8, 9, 10, 11, 12, 13, 14, 15);
}

// ---------------------------------------------------------------------------
// Persistent recurrent kernel: one workgroup per 16 batch rows, loops over T.
// ---------------------------------------------------------------------------
__global__ __launch_bounds__(THREADS, 1)
void ltc_kernel(const float* __restrict__ tx,
                const float* __restrict__ biasM_g,
                const float* __restrict__ biasA_g,
                const __bf16* __restrict__ WmF,
                const __bf16* __restrict__ WaF,
                float* __restrict__ out) {
    __shared__ float  stage [TB * NN];  // async-DMA staging for next x slab (fp32)
    __shared__ __bf16 x_bf  [TB * NN];  // current input (bf16: GEMM A + update)
    __shared__ float  mem_f [TB * NN];  // membrane state (f32, update precision)
    __shared__ float  b_f   [TB * NN];  // threshold state (f32)
    __shared__ __bf16 mem_bf[TB * NN];  // bf16 shadow of mem (GEMM A half)
    __shared__ __bf16 b_bf  [TB * NN];  // bf16 shadow of b   (GEMM A half)
    __shared__ __bf16 y_bf  [TB * NN];  // previous spikes (0/1, exact in bf16)

    const int tid  = threadIdx.x;
    const int lane = tid & 31;
    const int wave = tid >> 5;
    const int h    = lane >> 4;    // lane half: 0 or 1
    const int ln   = lane & 15;    // 0..15
    const int row0 = blockIdx.x * TB;
    const int nt0  = wave * TILES_PER_WAVE;
    const int m    = ln;           // A-matrix row owned by this lane

    // --- init state: mem = REST(0), b = THRESHOLD(0.3), y = 0 ---
    for (int i = tid; i < TB * NN; i += THREADS) {
        mem_f[i]  = 0.0f;
        b_f[i]    = 0.3f;
        mem_bf[i] = (__bf16)0.0f;
        b_bf[i]   = (__bf16)0.3f;
        y_bf[i]   = (__bf16)0.0f;
    }

    // --- per-tile bias values (constant across time) ---
    float biasM[TILES_PER_WAVE], biasA[TILES_PER_WAVE];
#pragma unroll
    for (int j = 0; j < TILES_PER_WAVE; ++j) {
        int n = (nt0 + j) * 16 + ln;
        biasM[j] = biasM_g[n];
        biasA[j] = biasA_g[n];
    }

    // --- prologue: DMA x(0) into stage ---
    async_slab_load(tx + (size_t)row0 * NN, stage, tid);
    wait_async0();
    __syncthreads();

    for (int t = 0; t < T_STEPS; ++t) {
        // ---- stage (fp32) -> x_bf ----
        for (int i = tid * 4; i < TB * NN; i += THREADS * 4) {
            float4 v = *(const float4*)(stage + i);
            x_bf[i + 0] = (__bf16)v.x;
            x_bf[i + 1] = (__bf16)v.y;
            x_bf[i + 2] = (__bf16)v.z;
            x_bf[i + 3] = (__bf16)v.w;
        }
        __syncthreads();   // x_bf ready; stage free to overwrite

        // ---- kick off DMA of next step's x, overlapped with the GEMMs ----
        if (t + 1 < T_STEPS)
            async_slab_load(tx + ((size_t)(t + 1) * BATCH + row0) * NN, stage, tid);

        // ---- dual GEMM: acc1 = [x|mem] @ Wm^T , acc2 = [x|b] @ Wa^T ----
        v8f acc1[TILES_PER_WAVE], acc2[TILES_PER_WAVE];
#pragma unroll
        for (int j = 0; j < TILES_PER_WAVE; ++j)
#pragma unroll
            for (int e = 0; e < 8; ++e) { acc1[j][e] = 0.0f; acc2[j][e] = 0.0f; }

        // K half 1: A = x (shared by both GEMMs)
#pragma unroll 2
        for (int kt = 0; kt < 16; ++kt) {
            v16bf a = a_frag(&x_bf[m * NN + kt * 32 + 8 * h]);
#pragma unroll
            for (int j = 0; j < TILES_PER_WAVE; ++j) {
                size_t foff = ((((size_t)(nt0 + j) * KT_TILES + kt) * 32) + lane) << 4;
                v16bf bm = *(const v16bf*)(WmF + foff);
                v16bf ba = *(const v16bf*)(WaF + foff);
                acc1[j] = __builtin_amdgcn_wmma_f32_16x16x32_bf16(
                    false, a, false, bm, (short)0, acc1[j], false, false);
                acc2[j] = __builtin_amdgcn_wmma_f32_16x16x32_bf16(
                    false, a, false, ba, (short)0, acc2[j], false, false);
            }
        }
        // K half 2: A1 = mem, A2 = b
#pragma unroll 2
        for (int kt = 16; kt < KT_TILES; ++kt) {
            const int off = m * NN + (kt - 16) * 32 + 8 * h;
            v16bf a1 = a_frag(&mem_bf[off]);
            v16bf a2 = a_frag(&b_bf[off]);
#pragma unroll
            for (int j = 0; j < TILES_PER_WAVE; ++j) {
                size_t foff = ((((size_t)(nt0 + j) * KT_TILES + kt) * 32) + lane) << 4;
                v16bf bm = *(const v16bf*)(WmF + foff);
                v16bf ba = *(const v16bf*)(WaF + foff);
                acc1[j] = __builtin_amdgcn_wmma_f32_16x16x32_bf16(
                    false, a1, false, bm, (short)0, acc1[j], false, false);
                acc2[j] = __builtin_amdgcn_wmma_f32_16x16x32_bf16(
                    false, a2, false, ba, (short)0, acc2[j], false, false);
            }
        }
        __syncthreads();   // all GEMM reads of state/x done before state writes

        // ---- elementwise LIF update (each (m,n) owned by exactly one lane) ----
#pragma unroll
        for (int j = 0; j < TILES_PER_WAVE; ++j) {
            const int n = (nt0 + j) * 16 + ln;
#pragma unroll
            for (int i = 0; i < 8; ++i) {
                const int mr  = i + 8 * h;          // C/D layout: M = i + 8*half
                const int idx = mr * NN + n;
                float alpha = 1.0f / (1.0f + __expf(-(acc1[j][i] + biasM[j])));
                float ro    = 1.0f / (1.0f + __expf(-(acc2[j][i] + biasA[j])));
                float yold  = (float)y_bf[idx];
                float bnew  = ro * b_f[idx] + (1.0f - ro) * yold;
                float Bthr  = 0.3f + 0.2f * bnew;
                float xv    = (float)x_bf[idx];
                float memv  = mem_f[idx];
                memv        = memv + (xv - memv) * alpha;
                float ynew  = (memv > Bthr) ? 1.0f : 0.0f;
                memv        = (1.0f - ynew) * memv;
                mem_f[idx]  = memv;
                mem_bf[idx] = (__bf16)memv;
                b_f[idx]    = bnew;
                b_bf[idx]   = (__bf16)bnew;
                y_bf[idx]   = (__bf16)ynew;
                out[((size_t)t * BATCH + row0 + mr) * NN + n] = ynew;
            }
        }
        wait_async0();     // this wave's DMA of x(t+1) has landed in LDS
        __syncthreads();   // all waves' DMA + state writes visible
    }
}

// ---------------------------------------------------------------------------
extern "C" void kernel_launch(void* const* d_in, const int* in_sizes, int n_in,
                              void* d_out, int out_size, void* d_ws, size_t ws_size,
                              hipStream_t stream) {
    const float* tx = (const float*)d_in[0];   // [T, BATCH, N] fp32
    const float* Wm = (const float*)d_in[1];   // [N, 2N] fp32
    const float* bM = (const float*)d_in[2];   // [N] fp32
    const float* Wa = (const float*)d_in[3];   // [N, 2N] fp32
    const float* bA = (const float*)d_in[4];   // [N] fp32

    __bf16* WmF = (__bf16*)d_ws;               // 1 MB bf16 fragments
    __bf16* WaF = WmF + (size_t)NN * KK;       // 1 MB bf16 fragments (2 MB total)

    const int total = NN * KK;                 // 524288 elements per matrix
    wconvert_kernel<<<total / 256, 256, 0, stream>>>(Wm, WmF);
    wconvert_kernel<<<total / 256, 256, 0, stream>>>(Wa, WaF);

    ltc_kernel<<<BATCH / TB, THREADS, 0, stream>>>(tx, bM, bA, WmF, WaF, (float*)d_out);
}
